// Causal_self_attention_77730318123444
// MI455X (gfx1250) — matrix operand order repaired and checked by reading, not compiled, not run on hardware
//
#include <hip/hip_runtime.h>
#include <hip/hip_bf16.h>

typedef __attribute__((ext_vector_type(16))) _Float16 v16h;
typedef __attribute__((ext_vector_type(8)))  _Float16 v8h;
typedef __attribute__((ext_vector_type(8)))  float    v8f;
typedef __attribute__((ext_vector_type(4)))  unsigned int u32x4;
typedef __attribute__((ext_vector_type(8)))  int      i32x8;
typedef __attribute__((ext_vector_type(4)))  int      i32x4;

#define BATCH 2
#define TSEQ  4096
#define CEMB  768
#define NHEAD 12
#define HDIM  64
#define N3C   2304

__device__ __forceinline__ v8f vzero8() { v8f r = {}; return r; }

__device__ __forceinline__ v8f wmma_f16f32(v16h a, v16h b, v8f c) {
  // (neg_a, A, neg_b, B, c_mod, C, reuse_a, reuse_b)
  return __builtin_amdgcn_wmma_f32_16x16x32_f16(false, a, false, b, (short)0, c,
                                                false, false);
}

__device__ __forceinline__ v16h join16(v8h lo, v8h hi) {
  v16h r;
#pragma unroll
  for (int i = 0; i < 8; ++i) { r[i] = lo[i]; r[i + 8] = hi[i]; }
  return r;
}

// A fragment (16x32 f16). rowp points at this lane's row (M = lane&15).
// Halves 0..7 at K = kbase..kbase+7, halves 8..15 at K = kbase+16..kbase+23,
// kbase = 8*(lane>>4)  (ISA 7.12.2).
__device__ __forceinline__ v16h load_a(const _Float16* rowp, int kbase) {
  v8h lo = *(const v8h*)(rowp + kbase);
  v8h hi = *(const v8h*)(rowp + kbase + 16);
  return join16(lo, hi);
}

// B fragment (32x16 f16). p points at this lane's column (N = lane&15) at K = 0,
// row stride ld; halves 0..7 at K = kbase..kbase+7, 8..15 at K = kbase+16..kbase+23.
__device__ __forceinline__ v16h load_b(const _Float16* p, size_t ld, int kbase) {
  v16h r;
#pragma unroll
  for (int i = 0; i < 8; ++i) {
    r[i] = p[(size_t)(kbase + i) * ld];
    r[i + 8] = p[(size_t)(kbase + 16 + i) * ld];
  }
  return r;
}

// One wave computes a 16x64 f32 tile: A fragment reused across 4 B tiles.
__device__ __forceinline__ void gemm_wave_16x64(
    const _Float16* __restrict__ A, int lda,
    const _Float16* __restrict__ B, int ldb,
    int K, int lane, v8f c[4]) {
#pragma unroll
  for (int i = 0; i < 4; ++i) c[i] = vzero8();
  const int row = lane & 15;
  const int kb  = 8 * (lane >> 4);
  const _Float16* arow = A + (size_t)row * lda;
#pragma unroll 2
  for (int k0 = 0; k0 < K; k0 += 32) {
    const _Float16* bp = B + (size_t)k0 * ldb + row;
    __builtin_prefetch(arow + k0 + 32, 0, 1);          // global_prefetch_b8
    __builtin_prefetch(bp + (size_t)32 * ldb, 0, 1);
    v16h a = load_a(arow + k0, kb);
    v16h b0 = load_b(bp, ldb, kb);
    v16h b1 = load_b(bp + 16, ldb, kb);
    v16h b2 = load_b(bp + 32, ldb, kb);
    v16h b3 = load_b(bp + 48, ldb, kb);
    c[0] = wmma_f16f32(a, b0, c[0]);
    c[1] = wmma_f16f32(a, b1, c[1]);
    c[2] = wmma_f16f32(a, b2, c[2]);
    c[3] = wmma_f16f32(a, b3, c[3]);
  }
}

// TDM: DMA a 2D f16 tile (rows x 64 cols, row stride HDIM elems) into LDS.
// D# per CDNA5 ISA (group0/group1), 2D -> groups 2/3 zero.
// This toolchain's builtin takes 6 args: (g0, g1, g2, g3, g4, cpol).
__device__ __forceinline__ void tdm_load_v_tile(unsigned lds_off,
                                                const _Float16* gsrc,
                                                unsigned rows) {
  unsigned long long ga = (unsigned long long)(uintptr_t)gsrc;
  u32x4 g0;
  g0[0] = 1u;                                         // count=1, user descriptor
  g0[1] = lds_off;                                    // lds_addr (bytes)
  g0[2] = (unsigned)ga;                               // global_addr[31:0]
  g0[3] = (unsigned)((ga >> 32) & 0x01ffffffu) | (2u << 30);  // addr[56:32] | type=2
  i32x8 g1;
  g1[0] = (int)(1u << 16);                            // data_size=1 (2 bytes)
  g1[1] = (int)((unsigned)HDIM << 16);                // tensor_dim0[15:0] @ bits63:48
  g1[2] = (int)((unsigned)TSEQ << 16);                // dim0 hi=0 | tensor_dim1[15:0]
  g1[3] = (int)((unsigned)HDIM << 16);                // dim1 hi=0 | tile_dim0=64
  g1[4] = (int)rows;                                  // tile_dim1 | tile_dim2=0
  g1[5] = HDIM;                                       // tensor_dim0_stride (elems)
  g1[6] = 0;
  g1[7] = 0;                                          // tensor_dim1_stride unused (2D)
  i32x4 gz4 = {0, 0, 0, 0};
  i32x8 gz8 = {0, 0, 0, 0, 0, 0, 0, 0};
  __builtin_amdgcn_tensor_load_to_lds(g0, g1, gz4, gz4, gz8, 0);
}

// ---------------------------------------------------------------------------
// fp32 -> fp16 conversion
__global__ void f32_to_f16_kernel(const float* __restrict__ in,
                                  _Float16* __restrict__ out, int n) {
  int i = blockIdx.x * blockDim.x + threadIdx.x;
  if (i < n) out[i] = (_Float16)in[i];
}

// ---------------------------------------------------------------------------
// QKV GEMM: [B*T, C] x [C, 3C] + bias, scattering into
//   k -> kT [B,H,D,T] (transposed so QK^T B-fragments load contiguously)
//   q -> q  [B,H,T,D] (pre-scaled by 1/sqrt(D))
//   v -> v  [B,H,T,D]
// Reference split order is (k, q, v).
__global__ __launch_bounds__(256) void qkv_gemm_kernel(
    const _Float16* __restrict__ X, const _Float16* __restrict__ W,
    const float* __restrict__ bias,
    _Float16* __restrict__ qb, _Float16* __restrict__ kT,
    _Float16* __restrict__ vb) {
  const int lane = threadIdx.x & 31;
  const int wave = threadIdx.x >> 5;              // 8 waves: 4 (M) x 2 (N)
  const int m0 = blockIdx.x * 64 + (wave >> 1) * 16;
  const int n0 = blockIdx.y * 128 + (wave & 1) * 64;
  v8f c[4];
  gemm_wave_16x64(X + (size_t)m0 * CEMB, CEMB, W + n0, N3C, CEMB, lane, c);

  const int mo = 8 * (lane >> 4);
#pragma unroll
  for (int ct = 0; ct < 4; ++ct) {
    const int col = n0 + ct * 16 + (lane & 15);
    const float bv = bias[col];
    const int sec = col / CEMB;                   // 0=k, 1=q, 2=v
    const int c2  = col - sec * CEMB;
    const int hh  = c2 >> 6;
    const int dd  = c2 & 63;
#pragma unroll
    for (int r = 0; r < 8; ++r) {
      const int rowi = m0 + r + mo;
      const int bb = rowi >> 12;                  // / TSEQ
      const int tt = rowi & (TSEQ - 1);
      const size_t bhh = (size_t)bb * NHEAD + hh;
      const float val = c[ct][r] + bv;
      if (sec == 0)
        kT[(bhh * HDIM + dd) * TSEQ + tt] = (_Float16)val;
      else if (sec == 1)
        qb[(bhh * TSEQ + tt) * HDIM + dd] = (_Float16)(val * 0.125f); // 1/sqrt(64)
      else
        vb[(bhh * TSEQ + tt) * HDIM + dd] = (_Float16)val;
    }
  }
}

// ---------------------------------------------------------------------------
// Causal flash attention. Block = 128 threads = 4 waves; wave w owns 16 query
// rows q0 = blockIdx.x*64 + w*16. Keys stream in steps of 32 with online
// softmax. V tiles arrive via the Tensor Data Mover into per-wave LDS; P is
// staged through LDS to convert C-layout -> A-layout.
#define VOFF(w) ((w) * 2048)            // 32 keys x 64 halfs per wave
#define POFF(w) (4 * 2048 + (w) * 512)  // 16 x 32 P staging per wave
__global__ __launch_bounds__(128) void flash_attn_kernel(
    const _Float16* __restrict__ Q, const _Float16* __restrict__ KT,
    const _Float16* __restrict__ V, _Float16* __restrict__ O) {
  __shared__ _Float16 smem[4 * 2048 + 4 * 512];
  const int lane = threadIdx.x & 31;
  const int wave = threadIdx.x >> 5;
  const int b = blockIdx.z, h = blockIdx.y;
  const int q0 = blockIdx.x * 64 + wave * 16;
  const size_t bh = (size_t)b * NHEAD + h;
  const _Float16* Qp  = Q  + bh * TSEQ * HDIM;
  const _Float16* KTp = KT + bh * HDIM * TSEQ;
  const _Float16* Vp  = V  + bh * TSEQ * HDIM;
  _Float16* Pl = &smem[POFF(wave)];
  _Float16* Vl = &smem[VOFF(wave)];
  // Flat LDS aperture carries the LDS byte offset in addr[31:0] (ISA 10.2).
  const unsigned v_lds_off = (unsigned)(uintptr_t)Vl;

  const int row = lane & 15;
  const int kb  = 8 * (lane >> 4);
  const int mo  = 8 * (lane >> 4);                // C-frag row offset

  const _Float16* qrow = Qp + (size_t)(q0 + row) * HDIM;
  const v16h qa0 = load_a(qrow, kb);              // d = 0..31 chunk
  const v16h qa1 = load_a(qrow + 32, kb);         // d = 32..63 chunk

  float mr[8], lr[8];
  v8f acc0 = vzero8(), acc1 = vzero8(), acc2 = vzero8(), acc3 = vzero8();
#pragma unroll
  for (int r = 0; r < 8; ++r) { mr[r] = -1e30f; lr[r] = 0.0f; }

  for (int j0 = 0; j0 <= q0 + 15; j0 += 32) {
    // ---- kick off the V-tile DMA for this iteration (overlaps with S) ----
    asm volatile("s_wait_dscnt 0" ::: "memory");  // prior ds reads done before overwrite
    tdm_load_v_tile(v_lds_off, Vp + (size_t)j0 * HDIM, 32);

    // ---- S = Q K^T for 32 keys (two 16-wide N chunks) ----
    v8f s0 = vzero8(), s1 = vzero8();
    {
      const _Float16* kp0 = KTp + j0 + row;                        // key = j0 + (lane&15), d = 0..31
      const _Float16* kp1 = KTp + (size_t)32 * TSEQ + j0 + row;    // d = 32..63
      s0 = wmma_f16f32(qa0, load_b(kp0, TSEQ, kb), s0);
      s0 = wmma_f16f32(qa1, load_b(kp1, TSEQ, kb), s0);
      s1 = wmma_f16f32(qa0, load_b(kp0 + 16, TSEQ, kb), s1);
      s1 = wmma_f16f32(qa1, load_b(kp1 + 16, TSEQ, kb), s1);
    }
    // ---- causal mask (tiles straddling the diagonal) ----
    if (j0 + 31 > q0) {
#pragma unroll
      for (int r = 0; r < 8; ++r) {
        const int qi = q0 + r + mo;
        const int kj = j0 + (lane & 15);
        if (kj > qi)      s0[r] = -1e30f;
        if (kj + 16 > qi) s1[r] = -1e30f;
      }
    }
    // ---- online softmax (row reductions across 16-lane halves) ----
    float pf0[8], pf1[8], alpha[8];
#pragma unroll
    for (int r = 0; r < 8; ++r) {
      float rm = fmaxf(s0[r], s1[r]);
      rm = fmaxf(rm, __shfl_xor(rm, 1));
      rm = fmaxf(rm, __shfl_xor(rm, 2));
      rm = fmaxf(rm, __shfl_xor(rm, 4));
      rm = fmaxf(rm, __shfl_xor(rm, 8));
      const float mn = fmaxf(mr[r], rm);
      const float a  = __expf(mr[r] - mn);
      const float p0 = __expf(s0[r] - mn);
      const float p1 = __expf(s1[r] - mn);
      float rs = p0 + p1;
      rs += __shfl_xor(rs, 1);
      rs += __shfl_xor(rs, 2);
      rs += __shfl_xor(rs, 4);
      rs += __shfl_xor(rs, 8);
      lr[r] = lr[r] * a + rs;
      mr[r] = mn;
      alpha[r] = a;
      pf0[r] = p0;
      pf1[r] = p1;
    }
#pragma unroll
    for (int r = 0; r < 8; ++r) {
      acc0[r] *= alpha[r]; acc1[r] *= alpha[r];
      acc2[r] *= alpha[r]; acc3[r] *= alpha[r];
    }
    // ---- stage P (C-layout) into LDS, reload as A fragment ----
#pragma unroll
    for (int r = 0; r < 8; ++r) {
      const int m = r + mo;
      Pl[m * 32 + (lane & 15)]      = (_Float16)pf0[r];
      Pl[m * 32 + 16 + (lane & 15)] = (_Float16)pf1[r];
    }
    asm volatile("s_wait_dscnt 0" ::: "memory");
    v16h pa;
    {
      const _Float16* lp = Pl + row * 32 + kb;
      v8h lo = *(const v8h*)(lp);
      v8h hi = *(const v8h*)(lp + 16);
      pa = join16(lo, hi);
    }
    // ---- O += P V from the TDM-filled LDS tile (4 d-chunks of 16) ----
    __builtin_amdgcn_s_wait_tensorcnt(0);
    {
      const _Float16* vl = Vl + row;                // d column = lane&15, key rows strided
      acc0 = wmma_f16f32(pa, load_b(vl, HDIM, kb), acc0);
      acc1 = wmma_f16f32(pa, load_b(vl + 16, HDIM, kb), acc1);
      acc2 = wmma_f16f32(pa, load_b(vl + 32, HDIM, kb), acc2);
      acc3 = wmma_f16f32(pa, load_b(vl + 48, HDIM, kb), acc3);
    }
    asm volatile("" ::: "memory");   // keep next iter's LDS traffic behind the reads
  }
  // ---- normalize + write attn output [B,T,C] (f16) ----
  _Float16* ob = O + (size_t)b * TSEQ * CEMB + (size_t)h * HDIM;
#pragma unroll
  for (int r = 0; r < 8; ++r) {
    const float inv = 1.0f / lr[r];
    const int qi = q0 + r + mo;
    _Float16* op = ob + (size_t)qi * CEMB + (lane & 15);
    op[0]  = (_Float16)(acc0[r] * inv);
    op[16] = (_Float16)(acc1[r] * inv);
    op[32] = (_Float16)(acc2[r] * inv);
    op[48] = (_Float16)(acc3[r] * inv);
  }
}

// ---------------------------------------------------------------------------
// Output projection: [B*T, C] x [C, C] + bias -> fp32 d_out
__global__ __launch_bounds__(256) void proj_gemm_kernel(
    const _Float16* __restrict__ A, const _Float16* __restrict__ W,
    const float* __restrict__ bias, float* __restrict__ out) {
  const int lane = threadIdx.x & 31;
  const int wave = threadIdx.x >> 5;
  const int m0 = blockIdx.x * 64 + (wave >> 1) * 16;
  const int n0 = blockIdx.y * 128 + (wave & 1) * 64;
  v8f c[4];
  gemm_wave_16x64(A + (size_t)m0 * CEMB, CEMB, W + n0, CEMB, CEMB, lane, c);
  const int mo = 8 * (lane >> 4);
#pragma unroll
  for (int ct = 0; ct < 4; ++ct) {
    const int col = n0 + ct * 16 + (lane & 15);
    const float bv = bias[col];
#pragma unroll
    for (int r = 0; r < 8; ++r) {
      const int rowi = m0 + r + mo;
      out[(size_t)rowi * CEMB + col] = c[ct][r] + bv;
    }
  }
}

// ---------------------------------------------------------------------------
extern "C" void kernel_launch(void* const* d_in, const int* in_sizes, int n_in,
                              void* d_out, int out_size, void* d_ws, size_t ws_size,
                              hipStream_t stream) {
  (void)in_sizes; (void)n_in; (void)out_size; (void)ws_size;
  const float* x      = (const float*)d_in[0];
  const float* W_attn = (const float*)d_in[1];
  const float* b_attn = (const float*)d_in[2];
  const float* W_proj = (const float*)d_in[3];
  const float* b_proj = (const float*)d_in[4];
  float* out = (float*)d_out;

  const int nX  = BATCH * TSEQ * CEMB;   // 6,291,456 (also = B*H*T*D)
  const int nWa = CEMB * N3C;            // 1,769,472
  const int nWp = CEMB * CEMB;           //   589,824

  char* ws = (char*)d_ws;
  size_t off = 0;
  auto alloc = [&](size_t bytes) {
    char* p = ws + off;
    off = (off + bytes + 255) & ~(size_t)255;
    return p;
  };
  _Float16* x16  = (_Float16*)alloc((size_t)nX * 2);
  _Float16* Wa16 = (_Float16*)alloc((size_t)nWa * 2);
  _Float16* Wp16 = (_Float16*)alloc((size_t)nWp * 2);
  _Float16* q16  = (_Float16*)alloc((size_t)nX * 2);
  _Float16* kT16 = (_Float16*)alloc((size_t)nX * 2);
  _Float16* v16  = (_Float16*)alloc((size_t)nX * 2);
  _Float16* ao16 = (_Float16*)alloc((size_t)nX * 2);

  const int thr = 256;
  f32_to_f16_kernel<<<(nX  + thr - 1) / thr, thr, 0, stream>>>(x, x16, nX);
  f32_to_f16_kernel<<<(nWa + thr - 1) / thr, thr, 0, stream>>>(W_attn, Wa16, nWa);
  f32_to_f16_kernel<<<(nWp + thr - 1) / thr, thr, 0, stream>>>(W_proj, Wp16, nWp);

  qkv_gemm_kernel<<<dim3(BATCH * TSEQ / 64, N3C / 128), 256, 0, stream>>>(
      x16, Wa16, b_attn, q16, kT16, v16);

  flash_attn_kernel<<<dim3(TSEQ / 64, NHEAD, BATCH), 128, 0, stream>>>(
      q16, kT16, v16, ao16);

  proj_gemm_kernel<<<dim3(BATCH * TSEQ / 64, CEMB / 128), 256, 0, stream>>>(
      ao16, Wp16, b_proj, out);
}
